// SparseRouter_30262339567884
// MI455X (gfx1250) — matrix operand order repaired and checked by reading, not compiled, not run on hardware
//
#include <hip/hip_runtime.h>
#include <stdint.h>

#define B_   32
#define J_   2048
#define H_   16
#define HD_  64
#define DQ_  1024
#define DM_  1024
#define DV_  1024
#define VHD_ 64
#define K_   8
#define IN_  2048

typedef __attribute__((ext_vector_type(16))) __bf16        v16bf;
typedef __attribute__((ext_vector_type(2)))  __bf16        v2bf;
typedef __attribute__((ext_vector_type(2)))  float         v2f;
typedef __attribute__((ext_vector_type(8)))  float         v8f;
typedef __attribute__((ext_vector_type(8)))  unsigned int  v8u;

#if defined(__has_builtin)
#if __has_builtin(__builtin_amdgcn_wmma_f32_16x16x4_f32)
#define USE_F32_WMMA 1
#endif
#if __has_builtin(__builtin_amdgcn_cvt_pk_bf16_f32)
#define HAVE_CVT_PK_BF16 1
#endif
#endif
#ifndef USE_F32_WMMA
#define USE_F32_WMMA 0
#endif
#ifndef HAVE_CVT_PK_BF16
#define HAVE_CVT_PK_BF16 0
#endif

// Padded LDS row strides: +16B so the 16 head rows land in distinct bank groups
// (64 banks x 4B = 256B period; stride*elt_size % 256 == 16).
#define QK_STRIDE_F32  1028
#define QK_STRIDE_BF16 1032

// Pack two f32 -> one dword of two bf16, low half = first K.
__device__ __forceinline__ unsigned int pack_bf16(float lo, float hi) {
#if HAVE_CVT_PK_BF16
  v2bf r = __builtin_amdgcn_cvt_pk_bf16_f32(lo, hi);
  return __builtin_bit_cast(unsigned int, r);
#else
  unsigned int ul = __builtin_bit_cast(unsigned int, lo);
  unsigned int uh = __builtin_bit_cast(unsigned int, hi);
  return ((uh + 0x8000u) & 0xFFFF0000u) | ((ul + 0x8000u) >> 16);
#endif
}

// ---------------------------------------------------------------------------
// K1: q[b, o] = u_t[b,:] . W_q[o,:]      (32 x 1024, tiny)
// ---------------------------------------------------------------------------
__global__ void __launch_bounds__(256)
k_qproj(const float* __restrict__ u_t, const float* __restrict__ W_q,
        float* __restrict__ q) {
  __shared__ float s_u[IN_];
  int b = blockIdx.x;
  for (int i = threadIdx.x; i < IN_; i += 256) s_u[i] = u_t[b * IN_ + i];
  __syncthreads();
  int o0 = threadIdx.x * 4;
  for (int r = 0; r < 4; ++r) {
    int o = o0 + r;
    const float* wr = W_q + (size_t)o * IN_;
    float acc = 0.f;
    for (int i = 0; i < IN_; i += 4) {
      float4 w4 = *(const float4*)(wr + i);
      acc += w4.x * s_u[i] + w4.y * s_u[i + 1] + w4.z * s_u[i + 2] + w4.w * s_u[i + 3];
    }
    q[b * DQ_ + o] = acc;
  }
}

// ---------------------------------------------------------------------------
// K2: fold q through W_k:  qk[b,h,m] = sum_d q[b,h,d] * W_k[h*64+d, m]
//     stored f32 (fp32-WMMA path) or bf16 (fallback path).
// ---------------------------------------------------------------------------
__global__ void __launch_bounds__(256)
k_qkfold(const float* __restrict__ q, const float* __restrict__ W_k,
         void* __restrict__ qk_out) {
  int idx = blockIdx.x * blockDim.x + threadIdx.x;  // [b][h][m]
  int m = idx & (DM_ - 1);
  int h = (idx >> 10) & (H_ - 1);
  int b = idx >> 14;
  const float* qv = q + b * DQ_ + h * HD_;
  const float* wk = W_k + (size_t)(h * HD_) * DM_ + m;
  float acc = 0.f;
#pragma unroll 8
  for (int d = 0; d < HD_; ++d) acc += qv[d] * wk[(size_t)d * DM_];
#if USE_F32_WMMA
  ((float*)qk_out)[idx] = acc;
#else
  unsigned int u = __builtin_bit_cast(unsigned int, acc);
  ((unsigned short*)qk_out)[idx] = (unsigned short)((u + 0x8000u) >> 16);
#endif
}

// ---------------------------------------------------------------------------
// K3: scores[b,h,j] = (memory[b,j,:] . qk[b,h,:]) * 0.125 - 0.01*ema[b,j]
//     Batched GEMM [J x DM] x [DM x H] per batch.
//     Preferred: v_wmma_f32_16x16x4_f32 (exact fp32, no cvt VALU).
//     Fallback:  v_wmma_f32_16x16x32_bf16 with on-the-fly bf16 pack.
//     One workgroup = 1 batch x 128 j rows; 8 waves x (16 j-rows x 16 heads).
// ---------------------------------------------------------------------------
__global__ void __launch_bounds__(256)
k_scores(const float* __restrict__ mem, const float* __restrict__ ema,
         const void* __restrict__ qk_v, float* __restrict__ scores) {
  int b  = blockIdx.x >> 4;
  int jt = blockIdx.x & 15;
  int lane = threadIdx.x & 31;
  int w    = threadIdx.x >> 5;
  int row  = lane & 15;   // A-matrix M index (lanes 0-15 / 16-31 both M=0..15)
  int seg  = lane >> 4;   // half-wave selects K sub-segments
  int j0   = jt * 128 + w * 16;
  const float* rowp = mem + ((size_t)b * J_ + j0 + row) * DM_;

  v8f c = {0.f, 0.f, 0.f, 0.f, 0.f, 0.f, 0.f, 0.f};

#if USE_F32_WMMA
  __shared__ float s_qk[H_ * QK_STRIDE_F32];  // ~66 KB, qk[b] f32, padded rows
  {
    const float* qk = (const float*)qk_v + (size_t)b * H_ * DM_;
    for (int i = threadIdx.x; i < (H_ * DM_) / 4; i += 256) {
      int h = i >> 8, m4 = i & 255;
      ((float4*)(s_qk + h * QK_STRIDE_F32))[m4] = ((const float4*)qk)[i];
    }
  }
  __syncthreads();
  const float* bq = s_qk + (lane & 15) * QK_STRIDE_F32;  // B: N(head) = lane%16
  // A 16x4 f32: lane<16 -> K {k0,k0+1}; lane>=16 -> K {k0+2,k0+3}. B 4x16 same split.
#pragma unroll 8
  for (int k0 = 0; k0 < DM_; k0 += 4) {
    float2 a2 = *(const float2*)(rowp + k0 + seg * 2);
    float2 b2 = *(const float2*)(bq + k0 + seg * 2);
    v2f A  = {a2.x, a2.y};
    v2f Bv = {b2.x, b2.y};
    c = __builtin_amdgcn_wmma_f32_16x16x4_f32(false, A, false, Bv,
                                              (short)0, c, false, false);
  }
#else
  __shared__ unsigned short s_qk[H_ * QK_STRIDE_BF16];  // ~33 KB, qk[b] bf16
  {
    const unsigned short* qk = (const unsigned short*)qk_v + (size_t)b * H_ * DM_;
    for (int i = threadIdx.x; i < (H_ * DM_) / 8; i += 256) {
      int h = i >> 7, c8 = i & 127;
      ((uint4*)(s_qk + h * QK_STRIDE_BF16))[c8] = ((const uint4*)qk)[i];
    }
  }
  __syncthreads();
  const unsigned short* bq = s_qk + (lane & 15) * QK_STRIDE_BF16;
  for (int k0 = 0; k0 < DM_; k0 += 32) {
    // A 16x32 bf16: lane<16 holds K {k0..k0+15}; lane>=16 holds K {k0+16..k0+31}
    float4 a0 = *(const float4*)(rowp + k0 + seg * 16);
    float4 a1 = *(const float4*)(rowp + k0 + seg * 16 + 4);
    float4 a2 = *(const float4*)(rowp + k0 + seg * 16 + 8);
    float4 a3 = *(const float4*)(rowp + k0 + seg * 16 + 12);
    v8u au;
    au[0] = pack_bf16(a0.x, a0.y); au[1] = pack_bf16(a0.z, a0.w);
    au[2] = pack_bf16(a1.x, a1.y); au[3] = pack_bf16(a1.z, a1.w);
    au[4] = pack_bf16(a2.x, a2.y); au[5] = pack_bf16(a2.z, a2.w);
    au[6] = pack_bf16(a3.x, a3.y); au[7] = pack_bf16(a3.z, a3.w);
    v16bf A = __builtin_bit_cast(v16bf, au);
    v8u bu = *(const v8u*)(bq + k0 + seg * 16);
    v16bf Bv = __builtin_bit_cast(v16bf, bu);
    c = __builtin_amdgcn_wmma_f32_16x16x32_bf16(false, A, false, Bv,
                                                (short)0, c, false, false);
  }
#endif

  // D layout: VGPR r -> M = r + seg*8, N = lane%16  => 8 consecutive j per lane
  int h   = lane & 15;
  int jr0 = j0 + seg * 8;
  float4 e0 = *(const float4*)(ema + b * J_ + jr0);
  float4 e1 = *(const float4*)(ema + b * J_ + jr0 + 4);
  float4 s0, s1;
  s0.x = c[0] * 0.125f - 0.01f * e0.x;
  s0.y = c[1] * 0.125f - 0.01f * e0.y;
  s0.z = c[2] * 0.125f - 0.01f * e0.z;
  s0.w = c[3] * 0.125f - 0.01f * e0.w;
  s1.x = c[4] * 0.125f - 0.01f * e1.x;
  s1.y = c[5] * 0.125f - 0.01f * e1.y;
  s1.z = c[6] * 0.125f - 0.01f * e1.z;
  s1.w = c[7] * 0.125f - 0.01f * e1.w;
  float* op = scores + ((size_t)(b * H_ + h)) * J_ + jr0;
  *(float4*)op       = s0;
  *(float4*)(op + 4) = s1;
}

// ---------------------------------------------------------------------------
// K4: top-8 + softmax per (b,h). One wave32 per row; 8 argmax passes with
//     cross-lane butterfly reduce (lowest index wins ties, like lax.top_k).
// ---------------------------------------------------------------------------
__global__ void __launch_bounds__(32)
k_topk(const float* __restrict__ scores, float* __restrict__ alpha_raw,
       int* __restrict__ idx_out) {
  int bh = blockIdx.x;
  const float* rowv = scores + (size_t)bh * J_;
  int lane = threadIdx.x;
  float selv[K_];
  int   seli[K_];
  for (int p = 0; p < K_; ++p) {
    float m = -3.4e38f;
    int mi = J_;
    for (int j = lane; j < J_; j += 32) {
      bool skip = false;
      for (int qq = 0; qq < p; ++qq) skip |= (seli[qq] == j);
      float v = rowv[j];
      if (!skip && (v > m || (v == m && j < mi))) { m = v; mi = j; }
    }
    for (int off = 16; off > 0; off >>= 1) {
      float om = __shfl_xor(m, off, 32);
      int   oi = __shfl_xor(mi, off, 32);
      if (om > m || (om == m && oi < mi)) { m = om; mi = oi; }
    }
    selv[p] = m;
    seli[p] = mi;
  }
  if (lane == 0) {
    float mx = selv[0];  // pass 0 is the max
    float e[K_], s = 0.f;
    for (int p = 0; p < K_; ++p) { e[p] = __expf(selv[p] - mx); s += e[p]; }
    float inv = 1.f / s;
    for (int p = 0; p < K_; ++p) {
      alpha_raw[bh * K_ + p] = e[p] * inv;
      idx_out[bh * K_ + p]   = seli[p];
    }
  }
}

// ---------------------------------------------------------------------------
// K5: slot-capacity cap + renormalize; writes final alpha + indices outputs.
// ---------------------------------------------------------------------------
__global__ void __launch_bounds__(128)
k_cap(const float* __restrict__ araw, const int* __restrict__ idx,
      float* __restrict__ afin, float* __restrict__ alpha_out,
      float* __restrict__ idx_out) {
  __shared__ int   s_idx[128];
  __shared__ float s_capped[128];
  int b = blockIdx.x, t = threadIdx.x;
  int   myidx = idx[b * 128 + t];
  float a     = araw[b * 128 + t];
  s_idx[t] = myidx;
  __syncthreads();
  int occ = 0;
  for (int j = 0; j < t; ++j) occ += (s_idx[j] == myidx);
  float capped = (occ < 4) ? a : 0.f;
  s_capped[t] = capped;
  __syncthreads();
  int h0 = t & ~7;
  float denom = 0.f;
  for (int k = 0; k < 8; ++k) denom += s_capped[h0 + k];
  float res = (denom > 0.f) ? capped / fmaxf(denom, 1e-9f) : a;
  afin[b * 128 + t]      = res;
  alpha_out[b * 128 + t] = res;
  idx_out[b * 128 + t]   = (float)myidx;
}

// ---------------------------------------------------------------------------
// K6: sparse readout. wsum[m] = sum_k alpha_k * memory[b, idx_k, m], then
//     r[b,h,d] = wsum . W_v[h*64+d,:]. Only 4096 memory rows touched total.
// ---------------------------------------------------------------------------
__global__ void __launch_bounds__(256)
k_read(const float* __restrict__ mem, const float* __restrict__ W_v,
       const float* __restrict__ afin, const int* __restrict__ idx,
       float* __restrict__ out) {
  __shared__ float s_w[DM_];
  __shared__ float s_a[K_];
  __shared__ int   s_i[K_];
  int bh = blockIdx.x;
  int b = bh >> 4, h = bh & 15;
  if (threadIdx.x < K_) {
    s_a[threadIdx.x] = afin[bh * K_ + threadIdx.x];
    s_i[threadIdx.x] = idx[bh * K_ + threadIdx.x];
  }
  __syncthreads();
  for (int m = threadIdx.x; m < DM_; m += 256) {
    float s = 0.f;
#pragma unroll
    for (int k = 0; k < K_; ++k)
      s += s_a[k] * mem[((size_t)b * J_ + s_i[k]) * DM_ + m];
    s_w[m] = s;
  }
  __syncthreads();
  if (threadIdx.x < VHD_) {
    int d = threadIdx.x;
    const float* wr = W_v + (size_t)(h * VHD_ + d) * DM_;
    float r = 0.f;
    for (int m = 0; m < DM_; m += 4) {
      float4 w4 = *(const float4*)(wr + m);
      r += w4.x * s_w[m] + w4.y * s_w[m + 1] + w4.z * s_w[m + 2] + w4.w * s_w[m + 3];
    }
    out[b * DV_ + h * VHD_ + d] = r;
  }
}

// ---------------------------------------------------------------------------
extern "C" void kernel_launch(void* const* d_in, const int* in_sizes, int n_in,
                              void* d_out, int out_size, void* d_ws, size_t ws_size,
                              hipStream_t stream) {
  const float* u_t = (const float*)d_in[0];
  const float* mem = (const float*)d_in[1];
  const float* ema = (const float*)d_in[2];
  const float* W_q = (const float*)d_in[3];
  const float* W_k = (const float*)d_in[4];
  const float* W_v = (const float*)d_in[5];
  float* out = (float*)d_out;
  // d_out layout (floats): r_t [0,32768) | alpha [32768,36864) | indices [36864,40960)

  // Workspace layout is path-independent (qk region sized for the f32 variant).
  char* ws = (char*)d_ws;
  float* q      = (float*)ws;  ws += (size_t)B_ * DQ_ * 4;           // 128 KB
  void*  qk     = (void*)ws;   ws += (size_t)B_ * H_ * DM_ * 4;      // 2 MB (f32 or bf16)
  float* scores = (float*)ws;  ws += (size_t)B_ * H_ * J_ * 4;       // 4 MB
  float* araw   = (float*)ws;  ws += (size_t)B_ * H_ * K_ * 4;
  int*   idx    = (int*)ws;    ws += (size_t)B_ * H_ * K_ * 4;
  float* afin   = (float*)ws;  ws += (size_t)B_ * H_ * K_ * 4;

  k_qproj <<<B_,                256, 0, stream>>>(u_t, W_q, q);
  k_qkfold<<<(B_*H_*DM_)/256,   256, 0, stream>>>(q, W_k, qk);
  k_scores<<<B_ * (J_/128),     256, 0, stream>>>(mem, ema, qk, scores);
  k_topk  <<<B_ * H_,            32, 0, stream>>>(scores, araw, idx);
  k_cap   <<<B_,                128, 0, stream>>>(araw, idx, afin, out + 32768, out + 36864);
  k_read  <<<B_ * H_,           256, 0, stream>>>(mem, W_v, afin, idx, out);
}
